// ResNext_69647189672449
// MI455X (gfx1250) — matrix operand order repaired and checked by reading, not compiled, not run on hardware
//
#include <hip/hip_runtime.h>
#include <hip/hip_bf16.h>

typedef __attribute__((ext_vector_type(16))) _Float16 v16h;
typedef __attribute__((ext_vector_type(8)))  float    v8f;

#define EP_NONE     0
#define EP_RELU     1
#define EP_ADD      2
#define EP_ADD_RELU 3
#define EP_RELU_ADD 4   // relu(conv) + resid  (res-block ordering)

// ---------------------------------------------------------------------------
// Prep kernels: fold eval-mode BN (+conv bias) into f16 weights / f32 bias.
// ---------------------------------------------------------------------------
__global__ void fold_conv(const float* __restrict__ w, const float* __restrict__ b,
                          const float* __restrict__ g, const float* __restrict__ bt,
                          const float* __restrict__ m, const float* __restrict__ v,
                          _Float16* __restrict__ wf, float* __restrict__ bf,
                          int OC, int Kt) {
  int i = blockIdx.x * blockDim.x + threadIdx.x;
  int total = OC * Kt;
  if (i < total) {
    int oc = i / Kt;
    float inv = g[oc] * rsqrtf(v[oc] + 1e-5f);
    wf[i] = (_Float16)(w[i] * inv);
  }
  if (i < OC) {
    float inv = g[i] * rsqrtf(v[i] + 1e-5f);
    bf[i] = b[i] * inv + bt[i] - m[i] * inv;
  }
}

// next-block branch merge: 32 branches of (1x1 conv 4->C) + per-(card,C) BN,
// summed over card  ==  one dense 128->C 1x1 conv.
__global__ void fold_w3_w(const float* __restrict__ w3,
                          const float* __restrict__ g, const float* __restrict__ bt,
                          const float* __restrict__ m, const float* __restrict__ v,
                          _Float16* __restrict__ wf, int C) {
  int i = blockIdx.x * blockDim.x + threadIdx.x;
  if (i >= C * 128) return;
  int c = i >> 7, k = i & 127, card = k >> 2, j = k & 3;
  int bi = card * C + c;
  float inv = g[bi] * rsqrtf(v[bi] + 1e-5f);
  wf[i] = (_Float16)(w3[bi * 4 + j] * inv);
}
__global__ void fold_w3_b(const float* __restrict__ b3,
                          const float* __restrict__ g, const float* __restrict__ bt,
                          const float* __restrict__ m, const float* __restrict__ v,
                          float* __restrict__ bf, int C) {
  int c = blockIdx.x * blockDim.x + threadIdx.x;
  if (c >= C) return;
  float s = 0.f;
  for (int card = 0; card < 32; ++card) {
    int bi = card * C + c;
    float inv = g[bi] * rsqrtf(v[bi] + 1e-5f);
    s += b3[bi] * inv + bt[bi] - m[bi] * inv;
  }
  bf[c] = s;
}

__global__ void f32_to_f16(const float* __restrict__ in, _Float16* __restrict__ out,
                           size_t n) {
  size_t i = blockIdx.x * (size_t)blockDim.x + threadIdx.x;
  if (i < n) out[i] = (_Float16)in[i];
}

// ---------------------------------------------------------------------------
// Implicit-GEMM conv via V_WMMA_F32_16X16X32_F16, templated on filter size so
// the im2col decompose uses constant-divisor arithmetic (no runtime idiv).
// Each wave computes a 16(oc) x 64(pixel) tile: one A fragment is reused
// across 4 B fragments / 4 accumulators per K-step (4 wmma per iteration).
// Fragment layouts per CDNA5 ISA 7.12.2 (wave32):
//   A (16x32 f16): lane L -> M=L&15; half h -> K = (h/8)*16 + (L>=16?8:0) + (h&7)
//   B (32x16 f16): lane L -> N=L&15; half h -> K = (h/8)*16 + (L>=16?8:0) + (h&7)
//   D (16x16 f32): vgpr r, lane L -> M = r + (L>=16?8:0), N = L&15
// 1x1 uses in this net always have pad==0 and (Hout-1)*stride < Hin, so the
// KH==KW==1 specialization needs no bounds checks at all.
// ---------------------------------------------------------------------------
template <int KH, int KW>
__global__ __launch_bounds__(128) void conv_wmma(
    const _Float16* __restrict__ in, const _Float16* __restrict__ wf,
    const float* __restrict__ bias, const _Float16* __restrict__ resid,
    _Float16* __restrict__ out,
    int N, int Cin, int Hin, int Win,
    int Cout, int Hout, int Wout,
    int stride, int pad, int ep) {
  constexpr int KK = KH * KW;
  const int lane = threadIdx.x & 31;
  const int wave = threadIdx.x >> 5;
  const int mt = blockIdx.y;
  const int tile0 = blockIdx.x * 16 + wave * 4;   // first of this wave's 4 tiles

  const int HW = Hout * Wout;
  const int HWin = Hin * Win;
  const int Npix = N * HW;
  const int Kt = Cin * KK;
  const int hi = (lane >= 16) ? 1 : 0;

  // per-N-tile pixel decomposition (4 pixels per lane)
  int nimg[4], sidx[4], ohs[4], ows[4];
  bool pv[4];
  #pragma unroll
  for (int i = 0; i < 4; ++i) {
    int p = (tile0 + i) * 16 + (lane & 15);
    pv[i] = p < Npix;
    int ni = pv[i] ? (p / HW) : 0;
    int s  = pv[i] ? (p % HW) : 0;
    nimg[i] = ni; sidx[i] = s;
    int oh = s / Wout, ow = s % Wout;
    ohs[i] = oh * stride - pad;       // top-left of receptive field
    ows[i] = ow * stride - pad;
  }

  const int mA = mt * 16 + (lane & 15);
  const bool mAvalid = mA < Cout;
  const _Float16* wrow = wf + (size_t)(mAvalid ? mA : 0) * Kt;

  v8f acc[4] = {};
  for (int k0 = 0; k0 < Kt; k0 += 32) {
    __builtin_prefetch(wrow + k0 + 64, 0, 1);
    v16h a;
    #pragma unroll
    for (int h = 0; h < 16; ++h) {
      int k = k0 + (h >> 3) * 16 + hi * 8 + (h & 7);
      a[h] = (mAvalid && k < Kt) ? wrow[k] : (_Float16)0.0f;
    }
    #pragma unroll
    for (int i = 0; i < 4; ++i) {
      v16h b;
      #pragma unroll
      for (int h = 0; h < 16; ++h) {
        int k = k0 + (h >> 3) * 16 + hi * 8 + (h & 7);
        _Float16 val = (_Float16)0.0f;
        if (pv[i] && k < Kt) {
          if constexpr (KH == 1 && KW == 1) {
            // pad==0 and strided sample always in bounds for this network
            val = in[((size_t)nimg[i] * Cin + k) * HWin +
                     (size_t)ohs[i] * Win + ows[i]];
          } else {
            int cin = k / KK;                 // constant-divisor
            int r = k - cin * KK;
            int kh = r / KW, kw = r - kh * KW;
            int ih = ohs[i] + kh, iw = ows[i] + kw;
            if (ih >= 0 && ih < Hin && iw >= 0 && iw < Win)
              val = in[((size_t)nimg[i] * Cin + cin) * HWin +
                       (size_t)ih * Win + iw];
          }
        }
        b[h] = val;
      }
      acc[i] = __builtin_amdgcn_wmma_f32_16x16x32_f16(false, a, false, b,
                                                      (short)0, acc[i], false, false);
    }
  }

  #pragma unroll
  for (int i = 0; i < 4; ++i) {
    if (!pv[i]) continue;
    #pragma unroll
    for (int r = 0; r < 8; ++r) {
      int oc = mt * 16 + r + hi * 8;
      if (oc >= Cout) continue;
      float v = acc[i][r] + bias[oc];
      size_t idx = ((size_t)nimg[i] * Cout + oc) * HW + sidx[i];
      if (ep == EP_RELU_ADD) v = fmaxf(v, 0.0f);
      if (ep >= EP_ADD) v += (float)resid[idx];
      if (ep == EP_RELU || ep == EP_ADD_RELU) v = fmaxf(v, 0.0f);
      out[idx] = (_Float16)v;
    }
  }
}

// ---------------------------------------------------------------------------
// Grouped 3x3 conv (128 ch, 32 groups x 4 ch), pad 1, stride 1, fused BN+ReLU.
// Only 36 MACs/output -> direct VALU kernel (K too small for WMMA tiles).
// ---------------------------------------------------------------------------
__global__ void gconv3x3(const _Float16* __restrict__ in,
                         const _Float16* __restrict__ wf,
                         const float* __restrict__ bias,
                         _Float16* __restrict__ out, int N, int H, int W) {
  size_t i = blockIdx.x * (size_t)blockDim.x + threadIdx.x;
  size_t total = (size_t)N * 128 * H * W;
  if (i >= total) return;
  int w = i % W; size_t t = i / W;
  int h = t % H; t /= H;
  int oc = t % 128; int n = (int)(t / 128);
  int g = oc >> 2;
  const _Float16* wp = wf + oc * 36;
  float acc = bias[oc];
  for (int j = 0; j < 4; ++j) {
    const _Float16* ip = in + ((size_t)n * 128 + g * 4 + j) * H * W;
    #pragma unroll
    for (int kh = 0; kh < 3; ++kh) {
      int ih = h - 1 + kh;
      if (ih < 0 || ih >= H) continue;
      #pragma unroll
      for (int kw = 0; kw < 3; ++kw) {
        int iw = w - 1 + kw;
        if (iw < 0 || iw >= W) continue;
        acc += (float)wp[j * 9 + kh * 3 + kw] * (float)ip[(size_t)ih * W + iw];
      }
    }
  }
  out[i] = (_Float16)fmaxf(acc, 0.0f);
}

__global__ void maxpool3x3s2(const _Float16* __restrict__ in,
                             _Float16* __restrict__ out,
                             int Hin, int Win, int Hout, int Wout, size_t total) {
  size_t i = blockIdx.x * (size_t)blockDim.x + threadIdx.x;
  if (i >= total) return;
  int ow = i % Wout; size_t t = i / Wout;
  int oh = t % Hout; t /= Hout;            // t = n*C + c
  const _Float16* p = in + t * (size_t)Hin * Win;
  float m = -3.0e38f;
  #pragma unroll
  for (int kh = 0; kh < 3; ++kh)
    #pragma unroll
    for (int kw = 0; kw < 3; ++kw)
      m = fmaxf(m, (float)p[(size_t)(oh * 2 + kh) * Win + (ow * 2 + kw)]);
  out[i] = (_Float16)m;
}

// global average pool over HW + FC (256 -> 1), one block per batch element
__global__ void avgpool_fc(const _Float16* __restrict__ in,
                           const float* __restrict__ fcw,
                           const float* __restrict__ fcb,
                           float* __restrict__ out, int C, int HW) {
  __shared__ float red[256];
  int b = blockIdx.x, c = threadIdx.x;
  const _Float16* p = in + ((size_t)b * C + c) * HW;
  float s = 0.f;
  for (int i = 0; i < HW; ++i) s += (float)p[i];
  red[c] = (s / (float)HW) * fcw[c];
  __syncthreads();
  for (int st = 128; st > 0; st >>= 1) {
    if (c < st) red[c] += red[c + st];
    __syncthreads();
  }
  if (c == 0) out[b] = red[0] + fcb[0];
}

// ---------------------------------------------------------------------------
extern "C" void kernel_launch(void* const* d_in, const int* in_sizes, int n_in,
                              void* d_out, int out_size, void* d_ws, size_t ws_size,
                              hipStream_t stream) {
  (void)in_sizes; (void)n_in; (void)out_size; (void)ws_size;
  const int B = 32;

  int idx = 0;
  auto F = [&]() { return (const float*)d_in[idx++]; };
  struct ConvP { const float *w, *b, *g, *bt, *m, *v; };
  auto getConv = [&]() { ConvP c; c.w=F(); c.b=F(); c.g=F(); c.bt=F(); c.m=F(); c.v=F(); return c; };
  struct ResP { ConvP c1, c2, cs; };
  auto getRes = [&]() { ResP r; r.c1=getConv(); r.c2=getConv(); r.cs=getConv(); return r; };
  struct NextP { ConvP c1, c2, c3; };
  auto getNext = [&]() { NextP n; n.c1=getConv(); n.c2=getConv(); n.c3=getConv(); return n; };

  const float* x = F();
  ConvP stem = getConv();
  ResP  r1 = getRes();
  NextP n1 = getNext(), n2 = getNext();
  ResP  r2 = getRes();
  NextP n3 = getNext(), n4 = getNext();
  ResP  r5 = getRes();
  NextP n5 = getNext(), n6 = getNext();
  const float* fcw = F();
  const float* fcb = F();

  // workspace allocator
  char* base = (char*)d_ws;
  size_t off = 0;
  auto alloc = [&](size_t bytes) -> void* {
    off = (off + 255) & ~(size_t)255;
    void* p = base + off; off += bytes; return p;
  };

  struct Folded { _Float16* w; float* b; };
  auto fold = [&](const ConvP& c, int OC, int Kt) {
    Folded f; f.w = (_Float16*)alloc((size_t)OC * Kt * 2); f.b = (float*)alloc((size_t)OC * 4);
    int total = OC * Kt;
    fold_conv<<<dim3((total + 255) / 256), dim3(256), 0, stream>>>(
        c.w, c.b, c.g, c.bt, c.m, c.v, f.w, f.b, OC, Kt);
    return f;
  };
  auto foldW3 = [&](const ConvP& c, int C) {
    Folded f; f.w = (_Float16*)alloc((size_t)C * 128 * 2); f.b = (float*)alloc((size_t)C * 4);
    fold_w3_w<<<dim3((C * 128 + 255) / 256), dim3(256), 0, stream>>>(c.w, c.g, c.bt, c.m, c.v, f.w, C);
    fold_w3_b<<<dim3((C + 255) / 256), dim3(256), 0, stream>>>(c.b, c.g, c.bt, c.m, c.v, f.b, C);
    return f;
  };

  Folded fstem = fold(stem, 64, 3 * 49);
  Folded fr1a = fold(r1.c1, 64, 64 * 25), fr1b = fold(r1.c2, 64, 64 * 25), fr1s = fold(r1.cs, 64, 64);
  Folded fn1a = fold(n1.c1, 128, 64),  fn1b = fold(n1.c2, 128, 36), fn1c = foldW3(n1.c3, 64);
  Folded fn2a = fold(n2.c1, 128, 64),  fn2b = fold(n2.c2, 128, 36), fn2c = foldW3(n2.c3, 64);
  Folded fr2a = fold(r2.c1, 128, 64 * 25), fr2b = fold(r2.c2, 128, 128 * 25), fr2s = fold(r2.cs, 128, 64);
  Folded fn3a = fold(n3.c1, 128, 128), fn3b = fold(n3.c2, 128, 36), fn3c = foldW3(n3.c3, 128);
  Folded fn4a = fold(n4.c1, 128, 128), fn4b = fold(n4.c2, 128, 36), fn4c = foldW3(n4.c3, 128);
  Folded fr5a = fold(r5.c1, 256, 128 * 25), fr5b = fold(r5.c2, 256, 256 * 25), fr5s = fold(r5.cs, 256, 128);
  Folded fn5a = fold(n5.c1, 128, 256), fn5b = fold(n5.c2, 128, 36), fn5c = foldW3(n5.c3, 256);
  Folded fn6a = fold(n6.c1, 128, 256), fn6b = fold(n6.c2, 128, 36), fn6c = foldW3(n6.c3, 256);

  // activation buffers (f16)
  size_t nX = (size_t)B * 3 * 224 * 224;
  _Float16* X16 = (_Float16*)alloc(nX * 2);
  size_t nS = (size_t)B * 64 * 109 * 109;               // stem out, reused for T1/T2
  _Float16* S = (_Float16*)alloc(nS * 2);
  size_t nP = (size_t)B * 64 * 54 * 54;                 // pool out, reused for T3
  _Float16* P = (_Float16*)alloc(nP * 2);
  _Float16* T1 = S;                                     // <= [32,128,54,54]
  _Float16* T2 = S + (size_t)B * 128 * 54 * 54;         // <= [32,128,54,54]
  _Float16* T3 = P;                                     // residual / in-place tensor

  auto conv = [&](const _Float16* in_, Folded f, const _Float16* resid, _Float16* out_,
                  int Cin, int Hin, int Win, int Cout, int Hout, int Wout,
                  int K, int stride, int pad, int ep) {
    int Npix = B * Hout * Wout;
    int ntiles = (Npix + 15) / 16;
    dim3 grid((ntiles + 15) / 16, (Cout + 15) / 16);   // 4 waves x 4 tiles per block
    if (K == 7)
      conv_wmma<7, 7><<<grid, dim3(128), 0, stream>>>(in_, f.w, f.b, resid, out_,
          B, Cin, Hin, Win, Cout, Hout, Wout, stride, pad, ep);
    else if (K == 5)
      conv_wmma<5, 5><<<grid, dim3(128), 0, stream>>>(in_, f.w, f.b, resid, out_,
          B, Cin, Hin, Win, Cout, Hout, Wout, stride, pad, ep);
    else
      conv_wmma<1, 1><<<grid, dim3(128), 0, stream>>>(in_, f.w, f.b, resid, out_,
          B, Cin, Hin, Win, Cout, Hout, Wout, stride, pad, ep);
  };

  // stem + maxpool
  f32_to_f16<<<dim3((unsigned)((nX + 255) / 256)), dim3(256), 0, stream>>>(x, X16, nX);
  conv(X16, fstem, nullptr, S, 3, 224, 224, 64, 109, 109, 7, 2, 0, EP_RELU);
  {
    size_t total = (size_t)B * 64 * 54 * 54;
    maxpool3x3s2<<<dim3((unsigned)((total + 255) / 256)), dim3(256), 0, stream>>>(
        S, P, 109, 109, 54, 54, total);
  }

  auto resBlk = [&](Folded a, Folded b, Folded s, int Cin, int Cout,
                    int Hin, int Hout, int stride) {
    conv(T3, a, nullptr, T1, Cin, Hin, Hin, Cout, Hout, Hout, 5, stride, 2, EP_RELU);
    conv(T3, s, nullptr, T2, Cin, Hin, Hin, Cout, Hout, Hout, 1, stride, 0, EP_NONE);
    conv(T1, b, T2, T3, Cout, Hout, Hout, Cout, Hout, Hout, 5, 1, 2, EP_RELU_ADD);
  };
  auto nextBlk = [&](Folded a, Folded b, Folded c, int C, int H) {
    conv(T3, a, nullptr, T1, C, H, H, 128, H, H, 1, 1, 0, EP_RELU);
    size_t total = (size_t)B * 128 * H * H;
    gconv3x3<<<dim3((unsigned)((total + 255) / 256)), dim3(256), 0, stream>>>(
        T1, b.w, b.b, T2, B, H, H);
    conv(T2, c, T3, T3, 128, H, H, C, H, H, 1, 1, 0, EP_ADD_RELU);  // in-place resid
  };

  // res1 (input is pooled tensor, which aliases T3)
  {
    conv(T3, fr1a, nullptr, T1, 64, 54, 54, 64, 54, 54, 5, 1, 2, EP_RELU);
    conv(T3, fr1s, nullptr, T2, 64, 54, 54, 64, 54, 54, 1, 1, 0, EP_NONE);
    conv(T1, fr1b, T2, T3, 64, 54, 54, 64, 54, 54, 5, 1, 2, EP_RELU_ADD);
  }
  nextBlk(fn1a, fn1b, fn1c, 64, 54);
  nextBlk(fn2a, fn2b, fn2c, 64, 54);
  resBlk(fr2a, fr2b, fr2s, 64, 128, 54, 27, 2);
  nextBlk(fn3a, fn3b, fn3c, 128, 27);
  nextBlk(fn4a, fn4b, fn4c, 128, 27);
  resBlk(fr5a, fr5b, fr5s, 128, 256, 27, 14, 2);
  nextBlk(fn5a, fn5b, fn5c, 256, 14);
  nextBlk(fn6a, fn6b, fn6c, 256, 14);

  avgpool_fc<<<dim3(32), dim3(256), 0, stream>>>(T3, fcw, fcb, (float*)d_out, 256, 14 * 14);
}
